// RoPEAttention_wMask_6906307412357
// MI455X (gfx1250) — matrix operand order repaired and checked by reading, not compiled, not run on hardware
//
#include <hip/hip_runtime.h>

// ---------------------------------------------------------------------------
// RoPE multi-head attention for MI455X (gfx1250), bf16 WMMA pipeline.
// ---------------------------------------------------------------------------

#define D_MODEL 512
#define HEADS   8
#define HD      64
#define SEQB    2048
#define BATCH   4
#define MTOT    (BATCH * SEQB)   // 8192

typedef __bf16 bf16;
typedef __attribute__((ext_vector_type(16))) __bf16 v16bf;
typedef __attribute__((ext_vector_type(8)))  __bf16 v8bf;
typedef __attribute__((ext_vector_type(8)))  float  v8f;

// float -> bf16 with round-to-nearest-even (pure bit ops, no libdevice)
__device__ __forceinline__ bf16 f2bf(float f) {
    union { float f; unsigned u; } x;
    x.f = f;
    unsigned r = (x.u + 0x7FFFu + ((x.u >> 16) & 1u)) >> 16;
    unsigned short h = (unsigned short)r;
    return __builtin_bit_cast(bf16, h);
}

// A fragment (16x32 bf16): lane holds row m = lane%16.
// elems j=0..7  -> K = khalf*8 + j        (khalf = lane/16)
// elems j=8..15 -> K = 16 + khalf*8 + (j-8)
// `row` points at A[m] + k_base; loads are two contiguous 16B chunks.
__device__ __forceinline__ v16bf load_a_frag(const bf16* row, int lane) {
    int kh = (lane >> 4) & 1;
    v8bf c0 = *reinterpret_cast<const v8bf*>(row + kh * 8);
    v8bf c1 = *reinterpret_cast<const v8bf*>(row + 16 + kh * 8);
    v16bf r;
#pragma unroll
    for (int j = 0; j < 8; ++j) { r[j] = c0[j]; r[j + 8] = c1[j]; }
    return r;
}

// B fragment (32x16 bf16): lane holds column n = lane%16, K-half = lane/16.
// elems e=0..7 -> K = khalf*8 + e; e=8..15 -> K = 16 + khalf*8 + (e-8).
// `row` points at W[n] + k_base.
__device__ __forceinline__ v16bf load_b_frag(const bf16* row, int lane) {
    int kh = (lane >> 4) & 1;
    v8bf c0 = *reinterpret_cast<const v8bf*>(row + kh * 8);
    v8bf c1 = *reinterpret_cast<const v8bf*>(row + 16 + kh * 8);
    v16bf r;
#pragma unroll
    for (int j = 0; j < 8; ++j) { r[j] = c0[j]; r[j + 8] = c1[j]; }
    return r;
}

__device__ __forceinline__ v8f wmma_bf16(v16bf a, v16bf b, v8f c) {
    return __builtin_amdgcn_wmma_f32_16x16x32_bf16(false, a, false, b,
                                                   (short)0, c, false, false);
}

// ---------------------------------------------------------------------------
// fp32 -> bf16 conversion
// ---------------------------------------------------------------------------
__global__ void cvt_bf16_kernel(const float* __restrict__ in,
                                bf16* __restrict__ out, int n) {
    int i = blockIdx.x * blockDim.x + threadIdx.x;
    if (i < n) out[i] = f2bf(in[i]);
}

// ---------------------------------------------------------------------------
// Gather cos/sin diagonals from the (2048,512,512) rotary matrix R.
// cs[(pos*256 + i)*2 + {0,1}] = {cos, sin} for pair i at position pos.
// cos = R[m, 2i, 2i]; sin = R[m, 2i+1, 2i].
// ---------------------------------------------------------------------------
__global__ void gather_cs_kernel(const float* __restrict__ R,
                                 float* __restrict__ cs, int n) {
    int i = blockIdx.x * blockDim.x + threadIdx.x;
    if (i >= n) return;
    int m = i >> 8;
    int j = i & 255;
    size_t base = ((size_t)m) << 18;  // m * 512 * 512
    cs[2 * i]     = R[base + (size_t)(2 * j) * 512 + (2 * j)];
    cs[2 * i + 1] = R[base + (size_t)(2 * j + 1) * 512 + (2 * j)];
}

// ---------------------------------------------------------------------------
// Generic WMMA GEMM: C[M x N] = A[M x K](bf16) @ W[N x K]^T (bf16), f32 acc.
// One wave computes a 32x64 tile: 2 A-fragments x 4 B-fragments -> 8 WMMAs
// per 32-wide K step (B-fragments reused across both M halves).
// Epilogue modes:
//   0: RoPE rotate (cos/sin table), store bf16 row-major [M x N]
//   1: +bias, store bf16 as [b][h][s][hd]
//   2: +bias, store bf16 as [b][h][hd][s]   (transposed V for P@V)
//   3: +bias, store f32 row-major [M x N]
// ---------------------------------------------------------------------------
__global__ __launch_bounds__(128) void gemm_wmma_kernel(
    const bf16* __restrict__ A, const bf16* __restrict__ W,
    const float* __restrict__ bias, const float* __restrict__ cs,
    void* __restrict__ outp, int Mtotal, int N, int K, int mode) {
    int lane = threadIdx.x & 31;
    int wid  = (blockIdx.x * blockDim.x + threadIdx.x) >> 5;
    int nblk64 = N >> 6;
    int mt = wid / nblk64;
    int nb = wid % nblk64;
    int row0 = mt << 5;               // 32 rows per wave
    if (row0 >= Mtotal) return;
    int col0 = nb << 6;               // 64 cols per wave
    int ln15 = lane & 15;

    const bf16* arow0 = A + (size_t)(row0 + ln15) * K;
    const bf16* arow1 = A + (size_t)(row0 + 16 + ln15) * K;

    v8f z = {0.f, 0.f, 0.f, 0.f, 0.f, 0.f, 0.f, 0.f};
    v8f acc[2][4] = {{z, z, z, z}, {z, z, z, z}};

    for (int kk = 0; kk < K; kk += 32) {
        v16bf a0 = load_a_frag(arow0 + kk, lane);
        v16bf a1 = load_a_frag(arow1 + kk, lane);
#pragma unroll
        for (int f = 0; f < 4; ++f) {
            const bf16* wrow = W + (size_t)(col0 + f * 16 + ln15) * K + kk;
            v16bf bfr = load_b_frag(wrow, lane);
            acc[0][f] = wmma_bf16(a0, bfr, acc[0][f]);
            acc[1][f] = wmma_bf16(a1, bfr, acc[1][f]);
        }
    }

    int mhalf = (lane >> 4) << 3;
#pragma unroll
    for (int a2 = 0; a2 < 2; ++a2) {
#pragma unroll
        for (int r = 0; r < 8; ++r) {
            int m = row0 + a2 * 16 + r + mhalf;
#pragma unroll
            for (int f = 0; f < 4; ++f) {
                int n = col0 + f * 16 + ln15;
                float v = acc[a2][f][r];
                if (mode == 0) {
                    // RoPE: pair (2i, 2i+1) sits in (even, odd) lane pair.
                    float p = __shfl_xor(v, 1);
                    int pos = m & (SEQB - 1);
                    int ci = n >> 1;
                    const float* c2 = cs + ((size_t)(pos << 8) + ci) * 2;
                    float c = c2[0], s = c2[1];
                    float rot = (lane & 1) ? (v * c - p * s) : (v * c + p * s);
                    ((bf16*)outp)[(size_t)m * N + n] = f2bf(rot);
                } else if (mode == 1) {
                    float val = v + bias[n];
                    int b = m >> 11, s = m & (SEQB - 1), h = n >> 6, hd = n & 63;
                    ((bf16*)outp)[(((size_t)(b * HEADS + h)) * SEQB + s) * HD +
                                  hd] = f2bf(val);
                } else if (mode == 2) {
                    float val = v + bias[n];
                    int b = m >> 11, s = m & (SEQB - 1), h = n >> 6, hd = n & 63;
                    ((bf16*)outp)[(((size_t)(b * HEADS + h)) * HD + hd) * SEQB +
                                  s] = f2bf(val);
                } else {
                    ((float*)outp)[(size_t)m * N + n] = v + bias[n];
                }
            }
        }
    }
}

// ---------------------------------------------------------------------------
// Flash attention with causal mask. grid = B*H*(SEQ/64) blocks, 128 threads.
// Wave w owns 16 q-rows; iterates 64-key blocks (4 score frags) with online
// softmax, so the cross-lane softmax reductions are amortized over 2x keys.
// qp/kp: [b][h][s][64] bf16 ; vT: [b][h][64][s] bf16 ; ao: [b][s][512] bf16.
// ---------------------------------------------------------------------------
__global__ __launch_bounds__(128) void flash_attn_kernel(
    const bf16* __restrict__ qp, const bf16* __restrict__ kp,
    const bf16* __restrict__ vT, bf16* __restrict__ ao) {
    __shared__ __attribute__((aligned(64))) bf16 Pt[4][16][64];

    int lane = threadIdx.x & 31;
    int wv = threadIdx.x >> 5;
    int bh = blockIdx.x >> 5;          // SEQ/64 = 32 q-tiles
    int qt = blockIdx.x & 31;
    int b = bh >> 3, h = bh & 7;
    const bf16* q = qp + (size_t)bh * SEQB * HD;
    const bf16* k = kp + (size_t)bh * SEQB * HD;
    const bf16* v = vT + (size_t)bh * HD * SEQB;
    int ln15 = lane & 15;
    int mhalf = (lane >> 4) << 3;
    int q0 = qt * 64 + wv * 16;

    v16bf qa0 = load_a_frag(q + (size_t)(q0 + ln15) * HD, lane);
    v16bf qa1 = load_a_frag(q + (size_t)(q0 + ln15) * HD + 32, lane);

    v8f z = {0.f, 0.f, 0.f, 0.f, 0.f, 0.f, 0.f, 0.f};
    v8f o[4] = {z, z, z, z};
    float mi[8], li[8];
#pragma unroll
    for (int r = 0; r < 8; ++r) { mi[r] = -3.0e38f; li[r] = 0.f; }

    int nkb = (q0 + 15) / 64 + 1;
    for (int kb = 0; kb < nkb; ++kb) {
        int kbase = kb * 64;
        // S = Q @ K^T : four 16x16 score frags (keys j*16..j*16+15 of block)
        v8f s[4] = {z, z, z, z};
#pragma unroll
        for (int j = 0; j < 4; ++j) {
            const bf16* krow = k + (size_t)(kbase + j * 16 + ln15) * HD;
            v16bf bk = load_b_frag(krow, lane);
            s[j] = wmma_bf16(qa0, bk, s[j]);
            bk = load_b_frag(krow + 32, lane);
            s[j] = wmma_bf16(qa1, bk, s[j]);
        }
        // causal mask + online softmax (row = r + mhalf, cols across lanes)
#pragma unroll
        for (int r = 0; r < 8; ++r) {
            int m = q0 + r + mhalf;
            float aj[4];
            float mx = -3.0e38f;
#pragma unroll
            for (int j = 0; j < 4; ++j) {
                int kj = kbase + j * 16 + ln15;
                aj[j] = (kj <= m) ? s[j][r] * 0.125f : -3.0e38f;
                mx = fmaxf(mx, aj[j]);
            }
            mx = fmaxf(mx, __shfl_xor(mx, 1));
            mx = fmaxf(mx, __shfl_xor(mx, 2));
            mx = fmaxf(mx, __shfl_xor(mx, 4));
            mx = fmaxf(mx, __shfl_xor(mx, 8));
            float mn = fmaxf(mi[r], mx);
            float rs = 0.f;
            float p[4];
#pragma unroll
            for (int j = 0; j < 4; ++j) {
                p[j] = __expf(aj[j] - mn);
                rs += p[j];
            }
            rs += __shfl_xor(rs, 1);
            rs += __shfl_xor(rs, 2);
            rs += __shfl_xor(rs, 4);
            rs += __shfl_xor(rs, 8);
            float corr = __expf(mi[r] - mn);
            li[r] = li[r] * corr + rs;
            mi[r] = mn;
#pragma unroll
            for (int f = 0; f < 4; ++f) o[f][r] = o[f][r] * corr;
#pragma unroll
            for (int j = 0; j < 4; ++j)
                Pt[wv][r + mhalf][j * 16 + ln15] = f2bf(p[j]);
        }
        // D-layout -> A-layout via per-wave LDS tile (LDS is in-order per wave)
        asm volatile("s_wait_dscnt 0x0" ::: "memory");
        v16bf pa0 = load_a_frag(&Pt[wv][ln15][0], lane);
        v16bf pa1 = load_a_frag(&Pt[wv][ln15][32], lane);
        // O += P @ V : V rows are hd (transposed), contiguous in key
#pragma unroll
        for (int f = 0; f < 4; ++f) {
            const bf16* vrow = v + (size_t)(f * 16 + ln15) * SEQB + kbase;
            v16bf vb = load_b_frag(vrow, lane);
            o[f] = wmma_bf16(pa0, vb, o[f]);
            vb = load_b_frag(vrow + 32, lane);
            o[f] = wmma_bf16(pa1, vb, o[f]);
        }
    }

#pragma unroll
    for (int r = 0; r < 8; ++r) {
        float inv = 1.0f / li[r];
        int m = q0 + r + mhalf;
#pragma unroll
        for (int f = 0; f < 4; ++f) {
            int col = h * HD + f * 16 + ln15;
            ao[((size_t)b * SEQB + m) * D_MODEL + col] = f2bf(o[f][r] * inv);
        }
    }
}

// ---------------------------------------------------------------------------
// Host-side launch
// ---------------------------------------------------------------------------
extern "C" void kernel_launch(void* const* d_in, const int* in_sizes, int n_in,
                              void* d_out, int out_size, void* d_ws, size_t ws_size,
                              hipStream_t stream) {
    (void)in_sizes; (void)n_in; (void)out_size; (void)ws_size;
    const float* x   = (const float*)d_in[0];
    const float* Wq  = (const float*)d_in[1];
    const float* Wk  = (const float*)d_in[2];
    const float* Wv  = (const float*)d_in[3];
    const float* R   = (const float*)d_in[4];
    const float* ipw = (const float*)d_in[5];
    const float* ipb = (const float*)d_in[6];
    const float* ow  = (const float*)d_in[7];
    const float* ob  = (const float*)d_in[8];

    char* ws = (char*)d_ws;
    size_t off = 0;
    auto alloc = [&](size_t bytes) -> void* {
        void* p = ws + off;
        off += (bytes + 255) & ~(size_t)255;
        return p;
    };

    bf16* xb   = (bf16*)alloc((size_t)MTOT * D_MODEL * 2);     // 8 MB
    bf16* wqb  = (bf16*)alloc((size_t)D_MODEL * D_MODEL * 2);
    bf16* wkb  = (bf16*)alloc((size_t)D_MODEL * D_MODEL * 2);
    bf16* wvb  = (bf16*)alloc((size_t)D_MODEL * D_MODEL * 2);
    bf16* ipwb = (bf16*)alloc((size_t)3 * D_MODEL * D_MODEL * 2);
    bf16* owb  = (bf16*)alloc((size_t)D_MODEL * D_MODEL * 2);
    float* cs  = (float*)alloc((size_t)SEQB * 256 * 2 * 4);    // 4 MB
    bf16* qr   = (bf16*)alloc((size_t)MTOT * D_MODEL * 2);     // RoPE'd q
    bf16* kr   = (bf16*)alloc((size_t)MTOT * D_MODEL * 2);
    bf16* vr   = (bf16*)alloc((size_t)MTOT * D_MODEL * 2);
    bf16* qpb  = (bf16*)alloc((size_t)MTOT * D_MODEL * 2);     // [b][h][s][hd]
    bf16* kpb  = (bf16*)alloc((size_t)MTOT * D_MODEL * 2);
    bf16* vtb  = (bf16*)alloc((size_t)MTOT * D_MODEL * 2);     // [b][h][hd][s]
    bf16* aob  = (bf16*)alloc((size_t)MTOT * D_MODEL * 2);     // attn out

    auto cvt = [&](const float* in, bf16* out, int n) {
        cvt_bf16_kernel<<<(n + 255) / 256, 256, 0, stream>>>(in, out, n);
    };
    cvt(x, xb, MTOT * D_MODEL);
    cvt(Wq, wqb, D_MODEL * D_MODEL);
    cvt(Wk, wkb, D_MODEL * D_MODEL);
    cvt(Wv, wvb, D_MODEL * D_MODEL);
    cvt(ipw, ipwb, 3 * D_MODEL * D_MODEL);
    cvt(ow, owb, D_MODEL * D_MODEL);

    {
        int n = SEQB * 256;
        gather_cs_kernel<<<(n + 255) / 256, 256, 0, stream>>>(R, cs, n);
    }

    // waves = (M/32) * (N/64) = 256 * 8 = 2048 -> 512 blocks of 128 threads
    const int gblocks = (MTOT / 32) * (D_MODEL / 64) / 4;

    // Stage 1: q/k/v = RoPE(x @ W^T)
    gemm_wmma_kernel<<<gblocks, 128, 0, stream>>>(xb, wqb, nullptr, cs, qr,
                                                  MTOT, D_MODEL, D_MODEL, 0);
    gemm_wmma_kernel<<<gblocks, 128, 0, stream>>>(xb, wkb, nullptr, cs, kr,
                                                  MTOT, D_MODEL, D_MODEL, 0);
    gemm_wmma_kernel<<<gblocks, 128, 0, stream>>>(xb, wvb, nullptr, cs, vr,
                                                  MTOT, D_MODEL, D_MODEL, 0);

    // Stage 2: head projections (+bias), V written transposed per head
    gemm_wmma_kernel<<<gblocks, 128, 0, stream>>>(qr, ipwb, ipb, nullptr, qpb,
                                                  MTOT, D_MODEL, D_MODEL, 1);
    gemm_wmma_kernel<<<gblocks, 128, 0, stream>>>(
        kr, ipwb + (size_t)D_MODEL * D_MODEL, ipb + D_MODEL, nullptr, kpb,
        MTOT, D_MODEL, D_MODEL, 1);
    gemm_wmma_kernel<<<gblocks, 128, 0, stream>>>(
        vr, ipwb + (size_t)2 * D_MODEL * D_MODEL, ipb + 2 * D_MODEL, nullptr,
        vtb, MTOT, D_MODEL, D_MODEL, 2);

    // Stage 3: causal flash attention
    flash_attn_kernel<<<BATCH * HEADS * (SEQB / 64), 128, 0, stream>>>(
        qpb, kpb, vtb, aob);

    // Stage 4: output projection -> fp32
    gemm_wmma_kernel<<<gblocks, 128, 0, stream>>>(aob, owb, ob, nullptr,
                                                  (float*)d_out,
                                                  MTOT, D_MODEL, D_MODEL, 3);
}
